// GAT_29119878266918
// MI455X (gfx1250) — matrix operand order repaired and checked by reading, not compiled, not run on hardware
//
#include <hip/hip_runtime.h>
#include <math.h>

typedef _Float16 f16;
typedef __attribute__((ext_vector_type(16))) _Float16 v16h;
typedef __attribute__((ext_vector_type(8)))  _Float16 v8h;
typedef __attribute__((ext_vector_type(8)))  float    v8f;

#define NEG_SLOPE 0.2f
#define GAT_EPS 1e-16f

// ---------------- conversion kernels ----------------
__global__ void k_f32_to_f16(const float* __restrict__ in, f16* __restrict__ out, int n) {
  int i = blockIdx.x * blockDim.x + threadIdx.x;
  if (i < n) out[i] = (f16)in[i];
}

// W row-major [K, Nout] -> Wt row-major [Nout, K] in f16
__global__ void k_transpose_cvt(const float* __restrict__ W, f16* __restrict__ Wt,
                                int K, int Nout) {
  int i = blockIdx.x * blockDim.x + threadIdx.x;
  if (i < K * Nout) {
    int k = i / Nout, n = i % Nout;
    Wt[(size_t)n * K + k] = (f16)W[i];
  }
}

// ---------------- WMMA GEMM: C[M,Nout] = A[M,K] * Bt[Nout,K]^T ----------------
// One wave computes one 16x16 tile. K multiple of 32, M multiple of 16, Nout multiple of 16.
__global__ void k_wmma_gemm(const f16* __restrict__ A, const f16* __restrict__ Bt,
                            float* __restrict__ C, int M, int Nout, int K) {
  const int lane  = threadIdx.x & 31;
  const int wave  = threadIdx.x >> 5;
  const bool hi   = lane >= 16;
  const int mTile = blockIdx.x;
  const int nTile = blockIdx.y * 4 + wave;
  if (nTile * 16 >= Nout) return;

  const int arow = mTile * 16 + (lane & 15);   // A: lane = row M
  const int bcol = nTile * 16 + (lane & 15);   // B: lane = col N
  const f16* Ap = A  + (size_t)arow * K;
  const f16* Bp = Bt + (size_t)bcol * K;

  v8f acc = {};
  for (int k0 = 0; k0 < K; k0 += 32) {
    // A 16x32 f16 layout: lanes 0-15 hold K={0..7,16..23}, lanes 16-31 K={8..15,24..31}
    const int ak = k0 + (hi ? 8 : 0);
    // B 32x16 f16 layout: same K mapping as A (lanes 0-15 K={0..7,16..23}, lanes 16-31 K={8..15,24..31})
    const int bk = k0 + (hi ? 8 : 0);
    v8h a0 = *(const v8h*)(Ap + ak);
    v8h a1 = *(const v8h*)(Ap + ak + 16);
    v8h b0 = *(const v8h*)(Bp + bk);
    v8h b1 = *(const v8h*)(Bp + bk + 16);
    v16h a, b;
#pragma unroll
    for (int i = 0; i < 8; ++i) { a[i] = a0[i]; a[8 + i] = a1[i]; b[i] = b0[i]; b[8 + i] = b1[i]; }
    acc = __builtin_amdgcn_wmma_f32_16x16x32_f16(false, a, false, b, (short)0, acc, false, false);
  }
  // C/D layout: lane%16 = col, VGPR i = row i (+8 for lanes 16-31)
  const int ocol  = nTile * 16 + (lane & 15);
  const int orow0 = mTile * 16 + (hi ? 8 : 0);
#pragma unroll
  for (int i = 0; i < 8; ++i)
    C[(size_t)(orow0 + i) * Nout + ocol] = acc[i];
}

// ---------------- attention scores: one wave32 per (node, head) row ----------------
__global__ void k_att_scores(const float* __restrict__ xw, const float* __restrict__ att_s,
                             const float* __restrict__ att_d, float* __restrict__ a_s,
                             float* __restrict__ a_d, int N, int H, int Cc) {
  const int lane = threadIdx.x & 31;
  const int wv   = (blockIdx.x * blockDim.x + threadIdx.x) >> 5;
  if (wv >= N * H) return;
  const int h = wv % H;
  const float* row = xw + (size_t)wv * Cc;        // xw is [N, H, Cc] contiguous
  const float* as  = att_s + h * Cc;
  const float* ad  = att_d + h * Cc;
  float s = 0.f, d = 0.f;
  for (int c = lane; c < Cc; c += 32) { float v = row[c]; s += v * as[c]; d += v * ad[c]; }
#pragma unroll
  for (int off = 16; off > 0; off >>= 1) { s += __shfl_xor(s, off); d += __shfl_xor(d, off); }
  if (lane == 0) { a_s[wv] = s; a_d[wv] = d; }
}

// order-preserving float<->uint mapping for atomic max
__device__ __forceinline__ unsigned f2ord(float f) {
  unsigned u = __float_as_uint(f);
  return (u & 0x80000000u) ? ~u : (u | 0x80000000u);
}
__device__ __forceinline__ float ord2f(unsigned u) {
  return (u & 0x80000000u) ? __uint_as_float(u & 0x7FFFFFFFu) : __uint_as_float(~u);
}

__device__ __forceinline__ void edge_sd(const int* __restrict__ srcA, const int* __restrict__ dstA,
                                        int e, int E, int& s, int& d) {
  if (e < E) { s = srcA[e]; d = dstA[e]; } else { s = e - E; d = e - E; }  // self loops
}

// pass 1: leaky-relu score, per-dst segment max (ordered-uint atomicMax)
__global__ void k_edge_max(const int* __restrict__ srcA, const int* __restrict__ dstA,
                           const float* __restrict__ a_s, const float* __restrict__ a_d,
                           float* __restrict__ ebuf, unsigned* __restrict__ mbuf,
                           int E, int N, int H) {
  int t = blockIdx.x * blockDim.x + threadIdx.x;
  if (t >= (E + N) * H) return;
  int e = t / H, h = t % H, s, d;
  edge_sd(srcA, dstA, e, E, s, d);
  float v = a_s[s * H + h] + a_d[d * H + h];
  v = (v > 0.f) ? v : NEG_SLOPE * v;
  ebuf[t] = v;
  atomicMax(mbuf + d * H + h, f2ord(v));
}

// pass 2: exp(e - max) and per-dst denom
__global__ void k_edge_exp(const int* __restrict__ srcA, const int* __restrict__ dstA,
                           float* __restrict__ ebuf, const unsigned* __restrict__ mbuf,
                           float* __restrict__ denom, int E, int N, int H) {
  int t = blockIdx.x * blockDim.x + threadIdx.x;
  if (t >= (E + N) * H) return;
  int e = t / H, h = t % H, s, d;
  edge_sd(srcA, dstA, e, E, s, d);
  float ex = __expf(ebuf[t] - ord2f(mbuf[d * H + h]));
  ebuf[t] = ex;
  atomicAdd(denom + d * H + h, ex);
}

// pass 3: out[dst,h,c] += alpha * xw[src,h,c]
// One thread per (edge, head, channel): channel fastest -> coalesced gather of
// xw rows and coalesced global_atomic_add_f32 into the destination row.
__global__ void k_edge_agg(const int* __restrict__ srcA, const int* __restrict__ dstA,
                           const float* __restrict__ ebuf, const float* __restrict__ denom,
                           const float* __restrict__ xw, float* __restrict__ out,
                           int E, int N, int H, int Cc) {
  long t = (long)blockIdx.x * blockDim.x + threadIdx.x;
  long total = (long)(E + N) * H * Cc;
  if (t >= total) return;
  int c  = (int)(t % Cc);
  long eh = t / Cc;
  int h  = (int)(eh % H);
  int e  = (int)(eh / H);
  int s, d;
  edge_sd(srcA, dstA, e, E, s, d);
  float alpha = ebuf[eh] / (denom[d * H + h] + GAT_EPS);
  float v = xw[(size_t)s * H * Cc + (size_t)h * Cc + c] * alpha;
  atomicAdd(out + (size_t)d * H * Cc + (size_t)h * Cc + c, v);
}

// bias + ELU, emit f32 and f16 copies for the next GEMM
__global__ void k_bias_elu(const float* __restrict__ in, const float* __restrict__ b,
                           float* __restrict__ h, f16* __restrict__ h16, int N, int F) {
  int i = blockIdx.x * blockDim.x + threadIdx.x;
  if (i >= N * F) return;
  int f = i % F;
  float v = in[i] + b[f];
  v = (v > 0.f) ? v : (__expf(v) - 1.f);
  h[i] = v; h16[i] = (f16)v;
}

// final bias + log_softmax over NC<=32 classes
__global__ void k_final(const float* __restrict__ out2, const float* __restrict__ b2,
                        float* __restrict__ y, int N, int NC) {
  int n = blockIdx.x * blockDim.x + threadIdx.x;
  if (n >= N) return;
  float v[32];
  float mx = -INFINITY;
  for (int c = 0; c < NC; ++c) { v[c] = out2[(size_t)n * NC + c] + b2[c]; mx = fmaxf(mx, v[c]); }
  float s = 0.f;
  for (int c = 0; c < NC; ++c) s += __expf(v[c] - mx);
  float lse = mx + logf(s);
  for (int c = 0; c < NC; ++c) y[(size_t)n * NC + c] = v[c] - lse;
}

extern "C" void kernel_launch(void* const* d_in, const int* in_sizes, int n_in,
                              void* d_out, int out_size, void* d_ws, size_t ws_size,
                              hipStream_t stream) {
  const int F = 512, H1 = 8, C1 = 64, HC = 512, NC = 32;
  const int N = in_sizes[0] / F;
  const int E = in_sizes[1] / 2;
  const int Etot = E + N;

  const float* x   = (const float*)d_in[0];
  const int*   ei  = (const int*)d_in[1];
  const float* W1  = (const float*)d_in[2];
  const float* as1 = (const float*)d_in[3];
  const float* ad1 = (const float*)d_in[4];
  const float* b1  = (const float*)d_in[5];
  const float* W2  = (const float*)d_in[6];
  const float* as2 = (const float*)d_in[7];
  const float* ad2 = (const float*)d_in[8];
  const float* b2  = (const float*)d_in[9];
  const int* srcA = ei;
  const int* dstA = ei + E;

  // workspace carve-out (256B aligned)
  char* wp = (char*)d_ws;
  auto carve = [&](size_t bytes) -> void* {
    void* p = wp; wp += (bytes + 255) & ~(size_t)255; return p;
  };
  f16*      xh   = (f16*)     carve((size_t)N * F  * 2);
  f16*      W1t  = (f16*)     carve((size_t)F * HC * 2);
  f16*      W2t  = (f16*)     carve((size_t)HC * NC * 2);
  float*    xw1  = (float*)   carve((size_t)N * HC * 4);
  float*    aS1  = (float*)   carve((size_t)N * H1 * 4);
  float*    aD1  = (float*)   carve((size_t)N * H1 * 4);
  unsigned* m1   = (unsigned*)carve((size_t)N * H1 * 4);
  float*    den1 = (float*)   carve((size_t)N * H1 * 4);
  float*    e1   = (float*)   carve((size_t)Etot * H1 * 4);
  float*    out1 = (float*)   carve((size_t)N * HC * 4);
  float*    hF   = (float*)   carve((size_t)N * HC * 4);
  f16*      h16  = (f16*)     carve((size_t)N * HC * 2);
  float*    xw2  = (float*)   carve((size_t)N * NC * 4);
  float*    aS2  = (float*)   carve((size_t)N * 4);
  float*    aD2  = (float*)   carve((size_t)N * 4);
  unsigned* m2   = (unsigned*)carve((size_t)N * 4);
  float*    den2 = (float*)   carve((size_t)N * 4);
  float*    e2   = (float*)   carve((size_t)Etot * 4);
  float*    out2 = (float*)   carve((size_t)N * NC * 4);

  auto cdiv = [](long a, long b) { return (int)((a + b - 1) / b); };
  const int T = 256;

  // ---- layer 1 feature transform ----
  k_f32_to_f16<<<cdiv((long)N * F, T), T, 0, stream>>>(x, xh, N * F);
  k_transpose_cvt<<<cdiv((long)F * HC, T), T, 0, stream>>>(W1, W1t, F, HC);
  k_transpose_cvt<<<cdiv((long)HC * NC, T), T, 0, stream>>>(W2, W2t, HC, NC);
  k_wmma_gemm<<<dim3(N / 16, (HC / 16 + 3) / 4), 128, 0, stream>>>(xh, W1t, xw1, N, HC, F);

  // ---- layer 1 attention ----
  k_att_scores<<<cdiv((long)N * H1 * 32, T), T, 0, stream>>>(xw1, as1, ad1, aS1, aD1, N, H1, C1);
  hipMemsetAsync(m1,   0, (size_t)N * H1 * 4, stream);   // 0 == ordered encoding of a very negative float
  hipMemsetAsync(den1, 0, (size_t)N * H1 * 4, stream);
  hipMemsetAsync(out1, 0, (size_t)N * HC * 4, stream);
  k_edge_max<<<cdiv((long)Etot * H1, T), T, 0, stream>>>(srcA, dstA, aS1, aD1, e1, m1, E, N, H1);
  k_edge_exp<<<cdiv((long)Etot * H1, T), T, 0, stream>>>(srcA, dstA, e1, m1, den1, E, N, H1);
  k_edge_agg<<<cdiv((long)Etot * H1 * C1, T), T, 0, stream>>>(srcA, dstA, e1, den1, xw1, out1, E, N, H1, C1);
  k_bias_elu<<<cdiv((long)N * HC, T), T, 0, stream>>>(out1, b1, hF, h16, N, HC);

  // ---- layer 2 feature transform ----
  k_wmma_gemm<<<dim3(N / 16, 1), 128, 0, stream>>>(h16, W2t, xw2, N, NC, HC);

  // ---- layer 2 attention (H=1, C=NC) ----
  k_att_scores<<<cdiv((long)N * 32, T), T, 0, stream>>>(xw2, as2, ad2, aS2, aD2, N, 1, NC);
  hipMemsetAsync(m2,   0, (size_t)N * 4, stream);
  hipMemsetAsync(den2, 0, (size_t)N * 4, stream);
  hipMemsetAsync(out2, 0, (size_t)N * NC * 4, stream);
  k_edge_max<<<cdiv((long)Etot, T), T, 0, stream>>>(srcA, dstA, aS2, aD2, e2, m2, E, N, 1);
  k_edge_exp<<<cdiv((long)Etot, T), T, 0, stream>>>(srcA, dstA, e2, m2, den2, E, N, 1);
  k_edge_agg<<<cdiv((long)Etot * NC, T), T, 0, stream>>>(srcA, dstA, e2, den2, xw2, out2, E, N, 1, NC);

  // ---- final: bias + log_softmax ----
  k_final<<<cdiv((long)N, T), T, 0, stream>>>(out2, b2, (float*)d_out, N, NC);

  (void)n_in; (void)out_size; (void)ws_size;
}
